// MiniTransformer_53566832116378
// MI455X (gfx1250) — matrix operand order repaired and checked by reading, not compiled, not run on hardware
//
#include <hip/hip_runtime.h>
#include <stdint.h>

// MI455X / gfx1250, wave32. One wave = 2 batch elements (16 rows x D=32).
// All GEMMs as V_WMMA_F32_16X16X32_F16 (f16 in, f32 accum).

typedef __attribute__((ext_vector_type(16))) _Float16 v16h;
typedef __attribute__((ext_vector_type(8)))  float    v8f;

#define WAVES_PER_BLOCK 4
#define TPB   (WAVES_PER_BLOCK * 32)
#define BPB   (WAVES_PER_BLOCK * 2)     // batch elems per block
#define NBAT  131072
#define NTOK  8
#define NDIM  32
#define NHID  64
#define NVOC  27

union ABu { v16h v; _Float16 h[16]; uint32_t u[8]; };

__device__ inline uint32_t pk2(float a, float b) {
  union { _Float16 h[2]; uint32_t u; } t;
  t.h[0] = (_Float16)a; t.h[1] = (_Float16)b;
  return t.u;
}

// Wave-private LDS RAW ordering: DS ops are in-order per wave; the memory
// clobber pins compiler ordering, s_wait_dscnt 0 pins hardware completion.
__device__ inline void lds_fence() {
  asm volatile("s_wait_dscnt 0" ::: "memory");
}

__device__ inline v8f wmma16(v16h a, v16h b, v8f c) {
  return __builtin_amdgcn_wmma_f32_16x16x32_f16(
      /*neg_a=*/false, a, /*neg_b=*/false, b,
      /*c_mod=*/(short)0, c, /*reuse_a=*/false, /*reuse_b=*/false);
}

// B operand (32x16 f16): staged in LDS as [tile][lane][j] packed dword pairs.
__device__ inline v16h load_b(const uint32_t* sW, int tile, int lane) {
  ABu b;
  const uint32_t* p = sW + tile * 256 + lane * 8;
#pragma unroll
  for (int j = 0; j < 8; ++j) b.u[j] = p[j];
  return b.v;
}

// A operand (16x32 f16) from f32 row-major LDS scratch (row stride ldr).
// A layout: lane<16 holds row=lane, K=h (0..15); lanes 16..31 same +16.
__device__ inline v16h load_a(const float* src, int ldr, int lane, int kmax) {
  ABu a;
  const int row  = lane & 15;
  const int koff = (lane >= 16) ? 16 : 0;
#pragma unroll
  for (int hh = 0; hh < 16; ++hh) {
    int k = hh + koff;
    float f = (k < kmax) ? src[row * ldr + k] : 0.0f;
    a.h[hh] = (_Float16)f;
  }
  return a.v;
}

// C/D layout (16x16 f32): lane<16 -> rows 0..7 col=lane; lane>=16 -> rows 8..15.
__device__ inline void store_cv(float* dst, int ldr, int lane, v8f c, int colbase) {
  const int n  = lane & 15;
  const int rb = (lane >= 16) ? 8 : 0;
#pragma unroll
  for (int i = 0; i < 8; ++i) dst[(rb + i) * ldr + colbase + n] = c[i];
}

__global__ __launch_bounds__(TPB) void mini_transformer_kernel(
    const int*   __restrict__ tokens,
    const float* __restrict__ tok_emb,
    const float* __restrict__ pos_emb,
    const float* __restrict__ Wq, const float* __restrict__ Wk,
    const float* __restrict__ Wv, const float* __restrict__ W1,
    const float* __restrict__ W2, const float* __restrict__ Wout,
    float* __restrict__ out)
{
  __shared__ __align__(16) uint32_t sW[16 * 256];           // 16 B-tiles, f16-pair packed
  __shared__ __align__(16) float    sS[WAVES_PER_BLOCK * 1536]; // 6KB scratch per wave

  const int tid  = threadIdx.x;
  const int lane = tid & 31;
  const int wv   = tid >> 5;
  const int keyn = lane & 15;
  const bool hi  = lane >= 16;

  // ---- Stage all weights into LDS in WMMA B-layout (f32 -> f16 pairs) ----
  for (int g = tid; g < 16 * 256; g += TPB) {
    const int tile = g >> 8;
    const int r    = g & 255;
    const int bl   = r >> 3;                 // B-layout lane slot 0..31
    const int j    = r & 7;                  // VGPR index 0..7
    const int n    = bl & 15;                // column within tile
    const int kh   = (bl & 16) ? 16 : 0;     // lanes 16..31 hold K=16..31
    const float* W; int ld, c0, kb, nv;
    switch (tile) {
      case 0:  W = Wq;   ld = 32; c0 = 0;  kb = 0;  nv = 32; break;
      case 1:  W = Wq;   ld = 32; c0 = 16; kb = 0;  nv = 32; break;
      case 2:  W = Wk;   ld = 32; c0 = 0;  kb = 0;  nv = 32; break;
      case 3:  W = Wk;   ld = 32; c0 = 16; kb = 0;  nv = 32; break;
      case 4:  W = Wv;   ld = 32; c0 = 0;  kb = 0;  nv = 32; break;
      case 5:  W = Wv;   ld = 32; c0 = 16; kb = 0;  nv = 32; break;
      case 6:  W = W1;   ld = 64; c0 = 0;  kb = 0;  nv = 64; break;
      case 7:  W = W1;   ld = 64; c0 = 16; kb = 0;  nv = 64; break;
      case 8:  W = W1;   ld = 64; c0 = 32; kb = 0;  nv = 64; break;
      case 9:  W = W1;   ld = 64; c0 = 48; kb = 0;  nv = 64; break;
      case 10: W = W2;   ld = 32; c0 = 0;  kb = 0;  nv = 32; break;
      case 11: W = W2;   ld = 32; c0 = 16; kb = 0;  nv = 32; break;
      case 12: W = W2;   ld = 32; c0 = 0;  kb = 32; nv = 32; break;
      case 13: W = W2;   ld = 32; c0 = 16; kb = 32; nv = 32; break;
      case 14: W = Wout; ld = 27; c0 = 0;  kb = 0;  nv = 27; break;
      default: W = Wout; ld = 27; c0 = 16; kb = 0;  nv = 27; break;
    }
    const int col = c0 + n;
    const int k0  = kb + kh + 2 * j;
    const float f0 = (col < nv) ? W[k0 * ld + col]       : 0.0f;
    const float f1 = (col < nv) ? W[(k0 + 1) * ld + col] : 0.0f;
    sW[g] = pk2(f0, f1);
  }
  __syncthreads();

  // ---- Per-wave scratch regions (reused as the pipeline advances) ----
  float* S   = sS + wv * 1536;
  float* Xs  = S;            // 16x32 f32 (then reused by Pst/Hst/Gst)
  float* Qst = S + 512;      // 16x32
  float* Kst = S + 1024;     // 16x32
  float* Pst = S;            // 16x16
  float* Hst = S;            // 16x32
  float* Y1s = S + 512;      // 16x64 (overwrites Q/K stacks)
  float* Gst = S;            // 16x32

  const int b0 = blockIdx.x * BPB + wv * 2;

  // ---- Gather x = tok_emb[token] + pos_emb  (2 lanes per row, float4) ----
  {
    const int r  = lane & 15;
    const int ch = (lane >> 4) * 16;
    const int bb = b0 + (r >> 3);
    const int t  = r & 7;
    const int tk = tokens[bb * NTOK + t];
    const float4* te = (const float4*)(tok_emb + tk * NDIM + ch);
    const float4* pe = (const float4*)(pos_emb + t * NDIM + ch);
    float4* xs = (float4*)(Xs + r * NDIM + ch);
#pragma unroll
    for (int q = 0; q < 4; ++q) {
      float4 a = te[q], p4 = pe[q];
      float4 o; o.x = a.x + p4.x; o.y = a.y + p4.y; o.z = a.z + p4.z; o.w = a.w + p4.w;
      xs[q] = o;
    }
  }
  lds_fence();

  // A-form of x, plus f32 residual copy in C-layout.
  const v16h a_x = load_a(Xs, 32, lane, 32);
  float xc0[8], xc1[8];
  {
    const int rb = hi ? 8 : 0;
#pragma unroll
    for (int i = 0; i < 8; ++i) {
      xc0[i] = Xs[(rb + i) * 32 + keyn];
      xc1[i] = Xs[(rb + i) * 32 + 16 + keyn];
    }
  }

  const v8f z = {};
  // ---- Q,K,V projections: 6 WMMAs ----
  v8f qc0 = wmma16(a_x, load_b(sW, 0, lane), z);
  v8f qc1 = wmma16(a_x, load_b(sW, 1, lane), z);
  v8f kc0 = wmma16(a_x, load_b(sW, 2, lane), z);
  v8f kc1 = wmma16(a_x, load_b(sW, 3, lane), z);
  v8f vc0 = wmma16(a_x, load_b(sW, 4, lane), z);
  v8f vc1 = wmma16(a_x, load_b(sW, 5, lane), z);

  store_cv(Qst, 32, lane, qc0, 0);  store_cv(Qst, 32, lane, qc1, 16);
  store_cv(Kst, 32, lane, kc0, 0);  store_cv(Kst, 32, lane, kc1, 16);
  lds_fence();

  // ---- scores = Q @ K^T (both batches stacked; off-diagonal masked later) ----
  ABu bkt;
  {
    const float* kr = Kst + keyn * 32 + (hi ? 16 : 0);   // row `keyn` of K-stack
#pragma unroll
    for (int j = 0; j < 8; ++j) bkt.u[j] = pk2(kr[2 * j], kr[2 * j + 1]);
  }
  const v16h a_q = load_a(Qst, 32, lane, 32);
  v8f scv = wmma16(a_q, bkt.v, z);

  // ---- masked causal softmax (VALU + 8-lane butterflies; no LDS) ----
  float p[8];
#pragma unroll
  for (int i = 0; i < 8; ++i) {
    // rows 0..7 (lanes<16) = batch b0, keys are cols 0..7; rows 8..15 = b0+1, keys 8..15
    const bool valid = hi ? (keyn >= 8 && (keyn - 8) <= i) : (keyn <= i);
    float s = valid ? scv[i] : -1e30f;
    float m = s;
    m = fmaxf(m, __shfl_xor(m, 1));
    m = fmaxf(m, __shfl_xor(m, 2));
    m = fmaxf(m, __shfl_xor(m, 4));
    const float e = __expf(s - m);
    float sum = e;
    sum += __shfl_xor(sum, 1);
    sum += __shfl_xor(sum, 2);
    sum += __shfl_xor(sum, 4);
    p[i] = valid ? e / sum : 0.0f;   // exact zeros kill cross-batch/non-causal terms
  }

  // P (16x16) -> A-layout via LDS (Xs region is dead now)
  {
    const int rb = hi ? 8 : 0;
#pragma unroll
    for (int i = 0; i < 8; ++i) Pst[(rb + i) * 16 + keyn] = p[i];
  }
  lds_fence();
  const v16h a_p = load_a(Pst, 16, lane, 16);   // K 16..31 zero-padded

  // V as B operand built from registers (shuffle rows 8..15 from upper half-wave)
  ABu bv0, bv1;
  {
    float o0[8], o1[8];
#pragma unroll
    for (int i = 0; i < 8; ++i) { o0[i] = __shfl_xor(vc0[i], 16); o1[i] = __shfl_xor(vc1[i], 16); }
#pragma unroll
    for (int j = 0; j < 4; ++j) {
      bv0.u[j]     = hi ? 0u : pk2(vc0[2 * j], vc0[2 * j + 1]); // B rows 0..7
      bv0.u[4 + j] = hi ? 0u : pk2(o0[2 * j],  o0[2 * j + 1]);  // B rows 8..15
      bv1.u[j]     = hi ? 0u : pk2(vc1[2 * j], vc1[2 * j + 1]);
      bv1.u[4 + j] = hi ? 0u : pk2(o1[2 * j],  o1[2 * j + 1]);
    }                                                           // B rows 16..31 = 0
  }
  v8f oc0 = wmma16(a_p, bv0.v, z);
  v8f oc1 = wmma16(a_p, bv1.v, z);

  // ---- residual + LayerNorm #1 (16-lane butterflies over D=32) ----
  float hc0[8], hc1[8];
#pragma unroll
  for (int i = 0; i < 8; ++i) {
    const float a = oc0[i] + xc0[i];
    const float b = oc1[i] + xc1[i];
    float s = a + b, s2 = a * a + b * b;
#pragma unroll
    for (int m = 1; m < 16; m <<= 1) { s += __shfl_xor(s, m); s2 += __shfl_xor(s2, m); }
    const float mu  = s * (1.0f / 32.0f);
    const float var = s2 * (1.0f / 32.0f) - mu * mu;
    const float inv = rsqrtf(var + 1e-5f);
    hc0[i] = (a - mu) * inv;
    hc1[i] = (b - mu) * inv;
  }

  // ---- FFN: relu(H @ W1) @ W2 : 8 WMMAs ----
  {
    const int rb = hi ? 8 : 0;
#pragma unroll
    for (int i = 0; i < 8; ++i) {
      Hst[(rb + i) * 32 + keyn]      = hc0[i];
      Hst[(rb + i) * 32 + 16 + keyn] = hc1[i];
    }
  }
  lds_fence();
  const v16h a_h = load_a(Hst, 32, lane, 32);
  v8f y1[4];
#pragma unroll
  for (int t4 = 0; t4 < 4; ++t4) {
    y1[t4] = wmma16(a_h, load_b(sW, 6 + t4, lane), z);
#pragma unroll
    for (int i = 0; i < 8; ++i) y1[t4][i] = fmaxf(y1[t4][i], 0.0f);
    store_cv(Y1s, 64, lane, y1[t4], t4 * 16);
  }
  lds_fence();
  const v16h a_y10 = load_a(Y1s, 64, lane, 32);        // K 0..31
  const v16h a_y11 = load_a(Y1s + 32, 64, lane, 32);   // K 32..63
  v8f y2c0 = wmma16(a_y10, load_b(sW, 10, lane), z);
  y2c0     = wmma16(a_y11, load_b(sW, 12, lane), y2c0);
  v8f y2c1 = wmma16(a_y10, load_b(sW, 11, lane), z);
  y2c1     = wmma16(a_y11, load_b(sW, 13, lane), y2c1);

  // ---- residual + LayerNorm #2 ----
  float gc0[8], gc1[8];
#pragma unroll
  for (int i = 0; i < 8; ++i) {
    const float a = y2c0[i] + hc0[i];
    const float b = y2c1[i] + hc1[i];
    float s = a + b, s2 = a * a + b * b;
#pragma unroll
    for (int m = 1; m < 16; m <<= 1) { s += __shfl_xor(s, m); s2 += __shfl_xor(s2, m); }
    const float mu  = s * (1.0f / 32.0f);
    const float var = s2 * (1.0f / 32.0f) - mu * mu;
    const float inv = rsqrtf(var + 1e-5f);
    gc0[i] = (a - mu) * inv;
    gc1[i] = (b - mu) * inv;
  }

  // ---- logits = G @ Wout (27 cols, zero-padded to 32): 2 WMMAs ----
  {
    const int rb = hi ? 8 : 0;
#pragma unroll
    for (int i = 0; i < 8; ++i) {
      Gst[(rb + i) * 32 + keyn]      = gc0[i];
      Gst[(rb + i) * 32 + 16 + keyn] = gc1[i];
    }
  }
  lds_fence();
  const v16h a_g = load_a(Gst, 32, lane, 32);
  v8f l0 = wmma16(a_g, load_b(sW, 14, lane), z);
  v8f l1 = wmma16(a_g, load_b(sW, 15, lane), z);

  // ---- store logits [b, t, 0..26] ----
  {
    const int rb = hi ? 8 : 0;
#pragma unroll
    for (int i = 0; i < 8; ++i) {
      const int row = rb + i;
      const int bb  = b0 + (row >> 3);
      const int t   = row & 7;
      const size_t base = ((size_t)bb * NTOK + t) * NVOC;
      out[base + keyn] = l0[i];
      if (keyn < NVOC - 16) out[base + 16 + keyn] = l1[i];
    }
  }
}

extern "C" void kernel_launch(void* const* d_in, const int* in_sizes, int n_in,
                              void* d_out, int out_size, void* d_ws, size_t ws_size,
                              hipStream_t stream) {
  (void)in_sizes; (void)n_in; (void)out_size; (void)d_ws; (void)ws_size;
  const int*   tokens  = (const int*)  d_in[0];
  const float* tok_emb = (const float*)d_in[1];
  const float* pos_emb = (const float*)d_in[2];
  const float* Wq   = (const float*)d_in[3];
  const float* Wk   = (const float*)d_in[4];
  const float* Wv   = (const float*)d_in[5];
  const float* W1   = (const float*)d_in[6];
  const float* W2   = (const float*)d_in[7];
  const float* Wout = (const float*)d_in[8];
  float* out = (float*)d_out;

  dim3 grid(NBAT / BPB), block(TPB);
  hipLaunchKernelGGL(mini_transformer_kernel, grid, block, 0, stream,
                     tokens, tok_emb, pos_emb, Wq, Wk, Wv, W1, W2, Wout, out);
}
